// SelectiveSSM_28930899706579
// MI455X (gfx1250) — matrix operand order repaired and checked by reading, not compiled, not run on hardware
//
#include <hip/hip_runtime.h>
#include <hip/hip_bf16.h>
#include <math.h>
#include <stdint.h>

// ---------------- problem constants ----------------
#define D_MODEL 1024
#define D_STATE 16
#define D_CONV  4
#define D_INNER 2048
#define DT_RANK 64
#define BATCH   2
#define SEQ     2048
#define BL      (BATCH * SEQ)            // 4096 rows
#define XDBL_N  (DT_RANK + 2 * D_STATE)  // 96
#define XDBL_NP 128                      // padded for GEMM

// ---------------- workspace layout (bytes) ----------------
#define OFF_XZ    ((size_t)0)                                   // (BL,4096) f32
#define OFF_XC    (OFF_XZ   + (size_t)BL * 4096 * 4)            // (BL,2048) f32
#define OFF_XDBL  (OFF_XC   + (size_t)BL * 2048 * 4)            // (BL,96)   f32
#define OFF_DLT   (OFF_XDBL + (size_t)BL * 96 * 4)              // (BL,2048) f32
#define OFF_XBF   (OFF_DLT  + (size_t)BL * 2048 * 4)            // (BL,1024) bf16
#define OFF_WIN   (OFF_XBF  + (size_t)BL * 1024 * 2)            // (4096,1024) bf16 (transposed)
#define OFF_WXP   (OFF_WIN  + (size_t)4096 * 1024 * 2)          // (128,2048) bf16 (transposed, padded)
#define OFF_WDT   (OFF_WXP  + (size_t)XDBL_NP * 2048 * 2)       // (2048,64) bf16 (transposed)
#define OFF_WOUT  (OFF_WDT  + (size_t)2048 * 64 * 2)            // (1024,2048) bf16 (transposed)
#define OFF_XCBF  (OFF_WOUT + (size_t)1024 * 2048 * 2)          // (BL,2048) bf16
#define OFF_XDBBF (OFF_XCBF + (size_t)BL * 2048 * 2)            // (BL,96)   bf16
#define OFF_YFBF  (OFF_XDBBF+ (size_t)BL * 96 * 2)              // (BL,2048) bf16
// total ~183 MB

// ---------------- WMMA types ----------------
typedef __attribute__((ext_vector_type(16))) __bf16 v16bf;
typedef __attribute__((ext_vector_type(8)))  float  v8f;

#define BMT 128
#define BNT 128
#define BKT 32
#define LDT 40   // bf16 row stride: 80B rows (16B aligned), 20-dword stride

// ---- CDNA5 async global->LDS staging (ASYNCcnt path) ----
#if __has_builtin(__builtin_amdgcn_global_load_async_to_lds_b128)
#define USE_ASYNC_LDS 1
#endif

#ifdef USE_ASYNC_LDS
typedef __attribute__((ext_vector_type(4))) int v4i;
typedef __attribute__((address_space(1))) v4i v4i_g;   // global
typedef __attribute__((address_space(3))) v4i v4i_l;   // LDS

static __device__ inline void async_cp16(const __bf16* g, __bf16* l) {
    // AS3 pointer = low 32 bits of the generic shared address (ISA aperture map)
    __builtin_amdgcn_global_load_async_to_lds_b128(
        (v4i_g*)(uintptr_t)g,
        (v4i_l*)(unsigned)(uintptr_t)l, 0, 0);
}
static __device__ inline void wait_async() {
#if __has_builtin(__builtin_amdgcn_s_wait_asynccnt)
    __builtin_amdgcn_s_wait_asynccnt(0);
#else
    asm volatile("s_wait_asynccnt 0x0" ::: "memory");
#endif
}
#endif

// A fragment, 16x32 bf16 (ISA 7.12.2): lanes 0-15 -> K {0..7,16..23},
// lanes 16-31 -> K {8..15,24..31}; VGPR v holds a K pair per dword.
static __device__ inline v16bf load_afrag(const __bf16* rowp, int khalf) {
    union { v16bf v; unsigned u[8]; } r;
    const unsigned* p = (const unsigned*)rowp;
    int o = khalf >> 1;
#pragma unroll
    for (int i = 0; i < 4; ++i) r.u[i] = p[o + i];
#pragma unroll
    for (int i = 0; i < 4; ++i) r.u[4 + i] = p[8 + o + i];
    return r.v;
}

// B fragment, 32x16 bf16: lanes 0-15 hold K {0..7,16..23}, lanes 16-31 hold K {8..15,24..31}.
static __device__ inline v16bf load_bfrag(const __bf16* rowp, int lanehi) {
    union { v16bf v; unsigned u[8]; } r;
    const unsigned* p = (const unsigned*)rowp;
    int o = lanehi * 4;
#pragma unroll
    for (int i = 0; i < 8; ++i) r.u[i] = p[o + (i & 3) + ((i >> 2) << 3)];
    return r.v;
}

// C = act( A(MxK bf16,lda) * Wt(NxK bf16,ldw)^T + bias ), f32 accumulate.
// act: 0=none, 1=softplus(x+bias[n]).  Cbf: optional bf16 mirror of C.
__global__ __launch_bounds__(256) void wmma_gemm_kernel(
    const __bf16* __restrict__ A, int lda,
    const __bf16* __restrict__ Wt, int ldw,
    float* __restrict__ C, int ldc,
    __bf16* __restrict__ Cbf,
    int M, int N, int K,
    const float* __restrict__ bias, int act)
{
    __shared__ __attribute__((aligned(16))) __bf16 As[BMT * LDT];
    __shared__ __attribute__((aligned(16))) __bf16 Bs[BNT * LDT];

    const int tid  = threadIdx.x;
    const int lane = tid & 31;
    const int wave = tid >> 5;
    const int m0 = blockIdx.x * BMT;
    const int n0 = blockIdx.y * BNT;
    const int wm0 = (wave & 3) * 32;        // wave tile: 32 rows x 64 cols
    const int wn0 = (wave >> 2) * 64;

    v8f acc[2][4] = {};

    const int lhi  = lane >> 4;
    const int llo  = lane & 15;
    const int khalf = lhi * 8;

    for (int k0 = 0; k0 < K; k0 += BKT) {
        // ---- stage A/B tiles: 128 rows x 32 bf16 = 128 x 64B, 16B chunks ----
#pragma unroll
        for (int h = 0; h < 2; ++h) {
            int c   = tid + h * 256;        // chunk 0..511
            int row = c >> 2;               // 0..127
            int col = (c & 3) * 8;          // bf16 element offset 0,8,16,24
            const __bf16* ga = A  + (size_t)(m0 + row) * lda + k0 + col;
            const __bf16* gb = Wt + (size_t)(n0 + row) * ldw + k0 + col;
            __bf16* la = As + row * LDT + col;
            __bf16* lb = Bs + row * LDT + col;
#ifdef USE_ASYNC_LDS
            async_cp16(ga, la);
            async_cp16(gb, lb);
#else
            *(uint4*)la = *(const uint4*)ga;
            *(uint4*)lb = *(const uint4*)gb;
#endif
        }
        if (k0 + BKT < K) {                 // global_prefetch_b8 for next K tile
            __builtin_prefetch((const void*)(A  + (size_t)(m0 + (tid >> 1)) * lda + k0 + BKT), 0, 1);
            __builtin_prefetch((const void*)(Wt + (size_t)(n0 + (tid >> 1)) * ldw + k0 + BKT), 0, 1);
        }
#ifdef USE_ASYNC_LDS
        wait_async();
#endif
        __syncthreads();

        v16bf afrag[2], bfrag[4];
#pragma unroll
        for (int i = 0; i < 2; ++i)
            afrag[i] = load_afrag(As + (wm0 + i * 16 + llo) * LDT, khalf);
#pragma unroll
        for (int j = 0; j < 4; ++j)
            bfrag[j] = load_bfrag(Bs + (wn0 + j * 16 + llo) * LDT, lhi);

#pragma unroll
        for (int i = 0; i < 2; ++i)
#pragma unroll
            for (int j = 0; j < 4; ++j)
                acc[i][j] = __builtin_amdgcn_wmma_f32_16x16x32_bf16(
                    false, afrag[i], false, bfrag[j],
                    (short)0, acc[i][j], false, false);
        __syncthreads();
    }

    // ---- epilogue: C/D layout -> lanes 0-15: M=r, lanes 16-31: M=8+r ----
#pragma unroll
    for (int i = 0; i < 2; ++i) {
#pragma unroll
        for (int j = 0; j < 4; ++j) {
            int gn = n0 + wn0 + j * 16 + llo;
            if (gn < N) {
                float bv = (act == 1) ? bias[gn] : 0.f;
#pragma unroll
                for (int r = 0; r < 8; ++r) {
                    int gm = m0 + wm0 + i * 16 + lhi * 8 + r;
                    float v = acc[i][j][r];
                    if (act == 1) {
                        v += bv;
                        v = (v > 20.f) ? v : log1pf(__expf(v));  // softplus
                    }
                    C[(size_t)gm * ldc + gn] = v;
                    if (Cbf) Cbf[(size_t)gm * ldc + gn] = (__bf16)v;
                }
            }
        }
    }
}

// f32 -> bf16 elementwise convert
__global__ __launch_bounds__(256) void cvt_bf16_kernel(
    const float* __restrict__ src, __bf16* __restrict__ dst, int n)
{
    int idx = blockIdx.x * 256 + threadIdx.x;
    if (idx < n) dst[idx] = (__bf16)src[idx];
}

// Wt[n][k] = bf16(W[k][n]), rows n >= N zero-padded to Npad
__global__ __launch_bounds__(256) void transpose_cvt_kernel(
    const float* __restrict__ W, __bf16* __restrict__ Wt,
    int K, int N, int Npad)
{
    int idx = blockIdx.x * 256 + threadIdx.x;
    if (idx >= K * Npad) return;
    int k = idx % K;
    int n = idx / K;
    Wt[(size_t)n * K + k] = (n < N) ? (__bf16)W[(size_t)k * N + n] : (__bf16)0.f;
}

// Depthwise causal conv (K=4) + bias + SiLU; writes f32 (scan) and bf16 (GEMM).
__global__ __launch_bounds__(256) void conv_silu_kernel(
    const float* __restrict__ xz,
    const float* __restrict__ conv_w,
    const float* __restrict__ conv_b,
    float* __restrict__ xc_out,
    __bf16* __restrict__ xc_bf)
{
    int idx = blockIdx.x * 256 + threadIdx.x;
    int d = idx & (D_INNER - 1);
    int l = (idx >> 11) & (SEQ - 1);
    int b = idx >> 22;
    float acc = conv_b[d];
#pragma unroll
    for (int k = 0; k < D_CONV; ++k) {
        int ll = l - (D_CONV - 1) + k;
        if (ll >= 0)
            acc += xz[((size_t)(b * SEQ + ll)) * (2 * D_INNER) + d] * conv_w[d * D_CONV + k];
    }
    float s = acc / (1.f + __expf(-acc));   // silu
    xc_out[idx] = s;
    xc_bf[idx] = (__bf16)s;
}

// Sequential selective scan, fused with +x*D and *silu(z); bf16 output for out_proj.
__global__ __launch_bounds__(256) void scan_kernel(
    const float* __restrict__ xz,     // z at col 2048+d
    const float* __restrict__ xc,     // conv+silu (BL,2048) f32
    const float* __restrict__ x_dbl,  // (BL,96): [64:80)=B, [80:96)=C
    const float* __restrict__ dlt,    // (BL,2048) softplus'd
    const float* __restrict__ A_log,
    const float* __restrict__ Dp,
    __bf16* __restrict__ yf)          // (BL,2048) bf16
{
    __shared__ float sBC[32];
    const int b = blockIdx.x >> 3;
    const int d = (blockIdx.x & 7) * 256 + threadIdx.x;

    float a[D_STATE], h[D_STATE];
#pragma unroll
    for (int n = 0; n < D_STATE; ++n) {
        a[n] = -__expf(A_log[d * D_STATE + n]);
        h[n] = 0.f;
    }
    const float Dd = Dp[d];

    for (int t = 0; t < SEQ; ++t) {
        size_t row = (size_t)b * SEQ + t;
        if (threadIdx.x < 32)
            sBC[threadIdx.x] = x_dbl[row * XDBL_N + DT_RANK + threadIdx.x];
        __syncthreads();

        float dt_v = dlt[row * D_INNER + d];
        float x_v  = xc[row * D_INNER + d];
        float dx   = dt_v * x_v;
        float y = 0.f;
#pragma unroll
        for (int n = 0; n < D_STATE; ++n) {
            float dA = __expf(dt_v * a[n]);
            h[n] = dA * h[n] + dx * sBC[n];
            y += h[n] * sBC[16 + n];
        }
        float zv = xz[row * (2 * D_INNER) + D_INNER + d];
        float sz = zv / (1.f + __expf(-zv));
        yf[row * D_INNER + d] = (__bf16)((y + x_v * Dd) * sz);
        __syncthreads();
    }
}

extern "C" void kernel_launch(void* const* d_in, const int* in_sizes, int n_in,
                              void* d_out, int out_size, void* d_ws, size_t ws_size,
                              hipStream_t stream) {
    (void)in_sizes; (void)n_in; (void)out_size; (void)ws_size;
    const float* x          = (const float*)d_in[0];
    const float* in_proj_w  = (const float*)d_in[1];
    const float* conv_w     = (const float*)d_in[2];
    const float* conv_b     = (const float*)d_in[3];
    const float* x_proj_w   = (const float*)d_in[4];
    const float* dt_proj_w  = (const float*)d_in[5];
    const float* dt_proj_b  = (const float*)d_in[6];
    const float* A_log      = (const float*)d_in[7];
    const float* Dp         = (const float*)d_in[8];
    const float* out_proj_w = (const float*)d_in[9];
    float* out = (float*)d_out;

    char* ws = (char*)d_ws;
    float*  xz      = (float*)(ws + OFF_XZ);
    float*  xcv     = (float*)(ws + OFF_XC);
    float*  x_dbl   = (float*)(ws + OFF_XDBL);
    float*  dlt     = (float*)(ws + OFF_DLT);
    __bf16* x_bf    = (__bf16*)(ws + OFF_XBF);
    __bf16* w_in_t  = (__bf16*)(ws + OFF_WIN);
    __bf16* w_xp_t  = (__bf16*)(ws + OFF_WXP);
    __bf16* w_dt_t  = (__bf16*)(ws + OFF_WDT);
    __bf16* w_out_t = (__bf16*)(ws + OFF_WOUT);
    __bf16* xc_bf   = (__bf16*)(ws + OFF_XCBF);
    __bf16* xdbl_bf = (__bf16*)(ws + OFF_XDBBF);
    __bf16* yf_bf   = (__bf16*)(ws + OFF_YFBF);

    // 0) precision/layout prep: x -> bf16; weights -> bf16 transposed (N-padded)
    cvt_bf16_kernel<<<(BL * D_MODEL) / 256, 256, 0, stream>>>(x, x_bf, BL * D_MODEL);
    transpose_cvt_kernel<<<(D_MODEL * 2 * D_INNER) / 256, 256, 0, stream>>>(
        in_proj_w, w_in_t, D_MODEL, 2 * D_INNER, 2 * D_INNER);
    transpose_cvt_kernel<<<(D_INNER * XDBL_NP) / 256, 256, 0, stream>>>(
        x_proj_w, w_xp_t, D_INNER, XDBL_N, XDBL_NP);
    transpose_cvt_kernel<<<(DT_RANK * D_INNER) / 256, 256, 0, stream>>>(
        dt_proj_w, w_dt_t, DT_RANK, D_INNER, D_INNER);
    transpose_cvt_kernel<<<(D_INNER * D_MODEL) / 256, 256, 0, stream>>>(
        out_proj_w, w_out_t, D_INNER, D_MODEL, D_MODEL);

    // 1) xz = x @ in_proj_w
    wmma_gemm_kernel<<<dim3(BL / BMT, (2 * D_INNER) / BNT), 256, 0, stream>>>(
        x_bf, D_MODEL, w_in_t, D_MODEL, xz, 2 * D_INNER, nullptr,
        BL, 2 * D_INNER, D_MODEL, nullptr, 0);

    // 2) depthwise causal conv + SiLU
    conv_silu_kernel<<<(BL * D_INNER) / 256, 256, 0, stream>>>(
        xz, conv_w, conv_b, xcv, xc_bf);

    // 3) x_dbl = xc @ x_proj_w (N=96, padded tile), + bf16 mirror for dt GEMM
    wmma_gemm_kernel<<<dim3(BL / BMT, 1), 256, 0, stream>>>(
        xc_bf, D_INNER, w_xp_t, D_INNER, x_dbl, XDBL_N, xdbl_bf,
        BL, XDBL_N, D_INNER, nullptr, 0);

    // 4) dlt = softplus(x_dbl[:, :64] @ dt_proj_w + dt_proj_b)
    wmma_gemm_kernel<<<dim3(BL / BMT, D_INNER / BNT), 256, 0, stream>>>(
        xdbl_bf, XDBL_N, w_dt_t, DT_RANK, dlt, D_INNER, nullptr,
        BL, D_INNER, DT_RANK, dt_proj_b, 1);

    // 5) selective scan (serial in t), fused +x*D and *silu(z), bf16 out
    scan_kernel<<<BATCH * (D_INNER / 256), 256, 0, stream>>>(
        xz, xcv, x_dbl, dlt, A_log, Dp, yf_bf);

    // 6) out = yf @ out_proj_w
    wmma_gemm_kernel<<<dim3(BL / BMT, D_MODEL / BNT), 256, 0, stream>>>(
        yf_bf, D_INNER, w_out_t, D_INNER, out, D_MODEL, nullptr,
        BL, D_MODEL, D_INNER, nullptr, 0);
}
